// TriangleAttentionStartingNode_86019605004398
// MI455X (gfx1250) — matrix operand order repaired and checked by reading, not compiled, not run on hardware
//
#include <hip/hip_runtime.h>
#include <hip/hip_bf16.h>

// ---------------- problem constants ----------------
constexpr int LDIM  = 320;
constexpr int CZd   = 128;
constexpr int NHEAD = 4;
constexpr int CHEAD = 32;
constexpr int HCd   = NHEAD * CHEAD;   // 128
constexpr int LL    = LDIM * LDIM;     // 102400
constexpr float SCALEf = 0.17677669529663687f;   // 1/sqrt(32)
constexpr float EPSf   = 1e-5f;

typedef _Float16 v16h __attribute__((ext_vector_type(16)));
typedef float    v8f  __attribute__((ext_vector_type(8)));
typedef _Float16 h8   __attribute__((ext_vector_type(8)));
typedef _Float16 h4   __attribute__((ext_vector_type(4)));
typedef int      v4i_t __attribute__((ext_vector_type(4)));
typedef unsigned int v4u_t __attribute__((ext_vector_type(4)));
typedef int      v8i_t __attribute__((ext_vector_type(8)));

// ---------------- async global -> LDS staging (gfx1250) ----------------
#if __has_builtin(__builtin_amdgcn_global_load_async_to_lds_b128) && \
    __has_builtin(__builtin_amdgcn_s_wait_asynccnt)
#define USE_ASYNC_LDS 1
typedef __attribute__((address_space(1))) v4i_t* gptr_v4i;   // global (__device__) int4*
typedef __attribute__((address_space(3))) v4i_t* sptr_v4i;   // LDS (__shared__) int4*
#else
#define USE_ASYNC_LDS 0
#endif

// copy 16 bytes (8 f16) from global to LDS, per lane
__device__ __forceinline__ void cp16(_Float16* l, const _Float16* g) {
#if USE_ASYNC_LDS
  void* gnc = (void*)const_cast<_Float16*>(g);
  __builtin_amdgcn_global_load_async_to_lds_b128(
      (gptr_v4i)gnc, (sptr_v4i)(void*)l,
      /*imm offset=*/0, /*cpol=*/0);
#else
  *(h8*)l = *(const h8*)g;
#endif
}

__device__ __forceinline__ void wait_staged() {
#if USE_ASYNC_LDS
  __builtin_amdgcn_s_wait_asynccnt(0);
#endif
  __syncthreads();
}

// ---------------- Tensor Data Mover staging (gfx1250 TDM) ----------------
#if __has_builtin(__builtin_amdgcn_tensor_load_to_lds) && \
    __has_builtin(__builtin_amdgcn_s_wait_tensorcnt)
#define USE_TDM 1
#else
#define USE_TDM 0
#endif

// LDS byte offset of a generic pointer into the workgroup's LDS allocation
__device__ __forceinline__ unsigned lds_off_of(const void* p) {
  return (unsigned)(unsigned long long)
      (__attribute__((address_space(3))) char*)(void*)p;
}

#if USE_TDM
// 2D TDM load: tile_h rows of tile_w 4-byte units, global rows strided by
// row_stride units; destination LDS is densely packed (tile_w per row).
// D# per CDNA5 ISA ch.8: group0 = {count, lds_addr, global_addr, type=2},
// group1 = {data_size, tensor_dim0/1, tile_dim0/1, tensor_dim0_stride}.
// Toolchain uses the 6-arg builtin: (g0, g1, g2, g3, g4, cpol).
__device__ __forceinline__ void tdm_load_2d(unsigned lds_off, const void* gsrc,
                                            unsigned tile_w, unsigned tile_h,
                                            unsigned row_stride) {
  const unsigned long long ga = (unsigned long long)gsrc;
  v4u_t g0;
  g0[0] = 1u;                                             // count=1, no gather
  g0[1] = lds_off;                                        // lds_addr [63:32]
  g0[2] = (unsigned)ga;                                   // global_addr [95:64]
  g0[3] = (unsigned)((ga >> 32) & 0x1FFFFFFu) | (2u << 30); // addr hi + type=2
  v8i_t g1;
  g1[0] = (int)(2u << 16);                 // data_size = 2 (4 bytes)
  g1[1] = (int)(tile_w << 16);             // tensor_dim0[15:0] @ [63:48]
  g1[2] = (int)(tile_h << 16);             // tensor_dim1[15:0] @ [95:80]
  g1[3] = (int)(tile_w << 16);             // tile_dim0 @ [127:112]
  g1[4] = (int)(tile_h & 0xFFFFu);         // tile_dim1 @ [143:128]
  g1[5] = (int)row_stride;                 // tensor_dim0_stride low32 @ [191:160]
  g1[6] = 0;
  g1[7] = 0;
  v4i_t z4;
  z4[0] = z4[1] = z4[2] = z4[3] = 0;       // groups 2/3 unused (2D tensor)
  v8i_t z8;
#pragma unroll
  for (int q = 0; q < 8; ++q) z8[q] = 0;   // extra group (unused)
  __builtin_amdgcn_tensor_load_to_lds(g0, g1, z4, z4, z8, /*cpol=*/0);
}
#endif

// ---------------- WMMA helpers ----------------
__device__ __forceinline__ v8f wmma_f16(const v16h a, const v16h b, const v8f c) {
  // D = A(16x32 f16) * B(32x16 f16) + C(16x16 f32)
  return __builtin_amdgcn_wmma_f32_16x16x32_f16(
      /*neg_a=*/false, a, /*neg_b=*/false, b,
      /*c_mod=*/(short)0, c, /*reuse_a=*/false, /*reuse_b=*/false);
}

__device__ __forceinline__ v8f zero8() {
  v8f z;
#pragma unroll
  for (int i = 0; i < 8; ++i) z[i] = 0.0f;
  return z;
}

__device__ __forceinline__ v16h cat8(const h8 lo, const h8 hi) {
  v16h r;
#pragma unroll
  for (int i = 0; i < 8; ++i) { r[i] = lo[i]; r[8 + i] = hi[i]; }
  return r;
}

// A-matrix 16x32 f16 fragment (ISA 7.12.2): lane m = lane&15;
// lanes 0-15 hold K {0..7, 16..23}, lanes 16-31 hold K {8..15, 24..31}.
__device__ __forceinline__ v16h load_a_frag(const _Float16* tile, int ld, int lane) {
  const int m    = lane & 15;
  const int koff = (lane & 16) ? 8 : 0;
  const _Float16* row = tile + m * ld;
  return cat8(*(const h8*)(row + koff), *(const h8*)(row + 16 + koff));
}

// B-matrix 32x16 f16 fragment: lane n = lane&15; lanes 0-15 hold K {0..7, 16..23},
// lanes 16-31 hold K {8..15, 24..31}. Storage must be K-contiguous:
// element (kr, n) at colbase[n*ld + kr] -> two aligned 16B LDS loads.
__device__ __forceinline__ v16h load_b_frag(const _Float16* colbase, int ld, int lane) {
  const int n     = lane & 15;
  const int kbase = (lane & 16) ? 8 : 0;
  const _Float16* p = colbase + n * ld + kbase;
  return cat8(*(const h8*)p, *(const h8*)(p + 16));
}

// ---------------- kernel 0: weights f32 -> f16, stored TRANSPOSED [n][k] ----------------
__global__ __launch_bounds__(256) void wconv_kernel(
    const float* __restrict__ Wq, const float* __restrict__ Wk,
    const float* __restrict__ Wv, const float* __restrict__ Wg,
    const float* __restrict__ Wo, _Float16* __restrict__ WcT) {
  const int w   = blockIdx.y;
  const int idx = blockIdx.x * 256 + threadIdx.x;   // 0..16383 enumerates (k, n)
  const int k = idx >> 7;
  const int n = idx & 127;
  const float* src = (w == 0) ? Wq : (w == 1) ? Wk : (w == 2) ? Wv : (w == 3) ? Wg : Wo;
  WcT[(size_t)w * (CZd * HCd) + n * CZd + k] = (_Float16)src[k * HCd + n];
}

// ---------------- kernel 1: LayerNorm (wave32 per pixel) ----------------
__global__ __launch_bounds__(256) void layernorm_kernel(
    const float* __restrict__ z, const float* __restrict__ gamma,
    const float* __restrict__ beta, _Float16* __restrict__ zln) {
  const int wave = threadIdx.x >> 5;
  const int lane = threadIdx.x & 31;
  const size_t pix = (size_t)blockIdx.x * 8 + wave;
  const float4 v = *(const float4*)(z + pix * CZd + lane * 4);
  float s = v.x + v.y + v.z + v.w;
#pragma unroll
  for (int m = 16; m >= 1; m >>= 1) s += __shfl_xor(s, m, 32);
  const float mu = s * (1.0f / 128.0f);
  const float dx = v.x - mu, dy = v.y - mu, dz_ = v.z - mu, dw = v.w - mu;
  float q = dx * dx + dy * dy + dz_ * dz_ + dw * dw;
#pragma unroll
  for (int m = 16; m >= 1; m >>= 1) q += __shfl_xor(q, m, 32);
  const float inv = rsqrtf(q * (1.0f / 128.0f) + EPSf);
  const float4 g = *(const float4*)(gamma + lane * 4);
  const float4 b = *(const float4*)(beta + lane * 4);
  h4 o;
  o[0] = (_Float16)(dx * inv * g.x + b.x);
  o[1] = (_Float16)(dy * inv * g.y + b.y);
  o[2] = (_Float16)(dz_ * inv * g.z + b.z);
  o[3] = (_Float16)(dw * inv * g.w + b.w);
  *(h4*)(zln + pix * CZd + lane * 4) = o;
}

// ---------------- kernel 2: attention bias B = z_ln @ Wb ----------------
__global__ __launch_bounds__(128) void bias_kernel(
    const _Float16* __restrict__ zln, const float* __restrict__ Wb,
    float* __restrict__ Bb) {
  __shared__ float red[128 * 4];
  const size_t pix = blockIdx.x;
  const int t = threadIdx.x;
  const float x = (float)zln[pix * CZd + t];
#pragma unroll
  for (int h = 0; h < 4; ++h) red[t * 4 + h] = x * Wb[t * 4 + h];
  __syncthreads();
  for (int s = 64; s >= 1; s >>= 1) {
    if (t < s) {
#pragma unroll
      for (int h = 0; h < 4; ++h) red[t * 4 + h] += red[(t + s) * 4 + h];
    }
    __syncthreads();
  }
  if (t < 4) Bb[pix * 4 + t] = red[t];
}

// ---------------- kernel 3: fused Q/K/V/G projections (WMMA GEMM) ----------------
// grid.x = LL/128 (M tiles), grid.y = 4 selects {Q,K,V,G}; block = 256 (8 waves).
// V is written TRANSPOSED per (i,h): Vt[((i*NHEAD+h)*CHEAD + c)*LDIM + k].
__global__ __launch_bounds__(256) void proj_gemm_kernel(
    const _Float16* __restrict__ zln, const _Float16* __restrict__ WcT,
    _Float16* __restrict__ Qb, _Float16* __restrict__ Kb,
    _Float16* __restrict__ Vt, _Float16* __restrict__ Gb) {
  extern __shared__ char smem[];
  _Float16* sA  = (_Float16*)smem;       // 128 x 128 activations, row-major [m][k]
  _Float16* sBT = sA + 128 * 128;        // 128 x 128 weight, K-contiguous [n][k]
  const int w = blockIdx.y;
  const size_t m0 = (size_t)blockIdx.x * 128;
  const int tid = threadIdx.x;
  const _Float16* gA = zln + m0 * HCd;
  const _Float16* gB = WcT + (size_t)w * CZd * HCd;
  for (int it = tid; it < 128 * 128 / 8; it += 256) {
    __builtin_prefetch(gA + 128 * 128 + it * 8, 0, 1);   // next M-tile -> global_prefetch_b8
    cp16(sA + it * 8, gA + it * 8);
    cp16(sBT + it * 8, gB + it * 8);
  }
  wait_staged();
  const int wave = tid >> 5, lane = tid & 31;
  const int mrow = wave * 16;
  v8f acc[8];
#pragma unroll
  for (int nt = 0; nt < 8; ++nt) acc[nt] = zero8();
#pragma unroll
  for (int ks = 0; ks < 4; ++ks) {
    const v16h a = load_a_frag(sA + mrow * HCd + ks * 32, HCd, lane);
#pragma unroll
    for (int nt = 0; nt < 8; ++nt) {
      const v16h b = load_b_frag(sBT + (nt * 16) * CZd + ks * 32, CZd, lane);
      acc[nt] = wmma_f16(a, b, acc[nt]);
    }
  }
  const int hi8 = (lane & 16) ? 8 : 0;
  const int n0  = lane & 15;
  if (w == 2) {
    // V transposed epilogue
#pragma unroll
    for (int nt = 0; nt < 8; ++nt) {
#pragma unroll
      for (int r = 0; r < 8; ++r) {
        const unsigned pix = (unsigned)(m0 + mrow + hi8 + r);
        const unsigned i = pix / LDIM;
        const unsigned k = pix - i * LDIM;
        const int col = nt * 16 + n0;
        Vt[(((size_t)i * NHEAD + (col >> 5)) * CHEAD + (col & 31)) * LDIM + k] =
            (_Float16)acc[nt][r];
      }
    }
  } else {
    _Float16* out = (w == 0) ? Qb : (w == 1) ? Kb : Gb;
#pragma unroll
    for (int nt = 0; nt < 8; ++nt) {
#pragma unroll
      for (int r = 0; r < 8; ++r) {
        float vv = acc[nt][r];
        if (w == 3) vv = 1.0f / (1.0f + __expf(-vv));   // sigmoid gate
        out[(m0 + mrow + hi8 + r) * HCd + nt * 16 + n0] = (_Float16)vv;
      }
    }
  }
}

// ---------------- kernel 4: triangle attention core ----------------
// grid = (LDIM/32 j-tiles, NHEAD, LDIM i-rows); block = 256 (8 waves).
// LDS: K(320x32 f16) | Vt(32x320 f16) | Q(32x32 f16) | P(32x320 f16) = 63,488 B
// K and V^T tiles are staged by the Tensor Data Mover (wave 0 issues, all wait).
__global__ __launch_bounds__(256) void tri_attn_kernel(
    const _Float16* __restrict__ Qb, const _Float16* __restrict__ Kb,
    const _Float16* __restrict__ Vt, const _Float16* __restrict__ Gb,
    const float* __restrict__ Bb, const unsigned char* __restrict__ res_mask,
    _Float16* __restrict__ Ob) {
  extern __shared__ char smem[];
  _Float16* sK  = (_Float16*)smem;              // [k=320][c=32]  (K-contiguous for B-frags)
  _Float16* sVT = sK + LDIM * CHEAD;            // [c=32][k=320]  (K-contiguous for B-frags)
  _Float16* sQ  = sVT + LDIM * CHEAD;           // [j=32][c=32]
  _Float16* sP  = sQ + 32 * CHEAD;              // [j=32][k=320] logits -> probs
  float* sRed = (float*)sK;                     // reused for split-K reduce (phase 4 only)

  const int jt = blockIdx.x * 32;
  const int h  = blockIdx.y;
  const int i  = blockIdx.z;
  const int tid = threadIdx.x;
  const int wave = tid >> 5, lane = tid & 31;

  // ---- stage K_i (TDM 2D), V_i^T (TDM 1D), Q_i (async) into LDS ----
#if USE_TDM
  if (wave == 0) {
    // K tile: 320 rows x 64B (16 dwords), global row stride 256B (64 dwords)
    tdm_load_2d(lds_off_of(sK),
                Kb + (size_t)i * LDIM * HCd + h * CHEAD,
                /*tile_w=*/16, /*tile_h=*/LDIM, /*row_stride=*/64);
    // V^T tile: contiguous 20480B (5120 dwords)
    tdm_load_2d(lds_off_of(sVT),
                Vt + ((size_t)i * NHEAD + h) * CHEAD * LDIM,
                /*tile_w=*/5120, /*tile_h=*/1, /*row_stride=*/5120);
  }
#else
  for (int it = tid; it < LDIM * CHEAD / 8; it += 256) {
    const int row = it >> 2;
    const int cc  = (it & 3) * 8;
    cp16(sK + row * CHEAD + cc,
         Kb + ((size_t)i * LDIM + row) * HCd + h * CHEAD + cc);
  }
  {
    const _Float16* gVt = Vt + ((size_t)i * NHEAD + h) * CHEAD * LDIM;
    for (int it = tid; it < LDIM * CHEAD / 8; it += 256)
      cp16(sVT + it * 8, gVt + it * 8);
  }
#endif
  for (int it = tid; it < 32 * CHEAD / 8; it += 256) {
    const int row = it >> 2;
    const int cc  = (it & 3) * 8;
    cp16(sQ + row * CHEAD + cc,
         Qb + ((size_t)i * LDIM + jt + row) * HCd + h * CHEAD + cc);
  }
#if USE_ASYNC_LDS
  __builtin_amdgcn_s_wait_asynccnt(0);
#endif
#if USE_TDM
  if (wave == 0) __builtin_amdgcn_s_wait_tensorcnt(0);
#endif
  __syncthreads();

  const int hi8 = (lane & 16) ? 8 : 0;
  const int n0  = lane & 15;

  // ---- phase 2: logits = Q K^T * scale + B, masked (2 m-tiles x 20 n-tiles) ----
  {
    const int mt = wave & 1;
    const v16h a = load_a_frag(sQ + mt * 16 * CHEAD, CHEAD, lane);
#pragma unroll
    for (int t = 0; t < 5; ++t) {
      const int nt = (wave >> 1) + 4 * t;
      // B fragment of K^T: element (c, n) = sK[(nt*16+n)*32 + c]
      const v16h b = load_b_frag(sK + (nt * 16) * CHEAD, CHEAD, lane);
      v8f c = wmma_f16(a, b, zero8());
      const int kg = nt * 16 + n0;
      const bool mk = res_mask[kg] != 0;
#pragma unroll
      for (int r = 0; r < 8; ++r) {
        const int jl = mt * 16 + hi8 + r;
        float lv = -1.0e30f;
        if (mk) lv = c[r] * SCALEf + Bb[((size_t)(jt + jl) * LDIM + kg) * NHEAD + h];
        sP[jl * LDIM + kg] = (_Float16)lv;
      }
    }
  }
  __syncthreads();

  // ---- phase 3: softmax over k (4 rows per wave, lane-parallel) ----
#pragma unroll
  for (int rr = 0; rr < 4; ++rr) {
    const int row = wave * 4 + rr;
    float mx = -1.0e30f;
    for (int k = lane; k < LDIM; k += 32) mx = fmaxf(mx, (float)sP[row * LDIM + k]);
#pragma unroll
    for (int m = 16; m >= 1; m >>= 1) mx = fmaxf(mx, __shfl_xor(mx, m, 32));
    float sum = 0.0f;
    float ebuf[10];
#pragma unroll
    for (int t = 0; t < 10; ++t) {
      const float e = __expf((float)sP[row * LDIM + lane + t * 32] - mx);
      ebuf[t] = e;
      sum += e;
    }
#pragma unroll
    for (int m = 16; m >= 1; m >>= 1) sum += __shfl_xor(sum, m, 32);
    const float inv = (sum > 0.0f) ? (1.0f / sum) : 0.0f;
#pragma unroll
    for (int t = 0; t < 10; ++t)
      sP[row * LDIM + lane + t * 32] = (_Float16)(ebuf[t] * inv);
  }
  __syncthreads();

  // ---- phase 4: out = P @ V ; split-K across wave pairs ----
  const int mt2 = wave & 1;
  const int nt2 = (wave >> 1) & 1;
  const int kh  = wave >> 2;          // 0: k=0..159, 1: k=160..319
  v8f o = zero8();
#pragma unroll
  for (int t = 0; t < 5; ++t) {
    const int ks = kh * 5 + t;
    const v16h a = load_a_frag(sP + mt2 * 16 * LDIM + ks * 32, LDIM, lane);
    // B fragment of V: element (kr, c) = sVT[(nt2*16+c)*320 + ks*32 + kr]
    const v16h b = load_b_frag(sVT + (nt2 * 16) * LDIM + ks * 32, LDIM, lane);
    o = wmma_f16(a, b, o);
  }
  __syncthreads();                    // K-tile no longer read; safe to reuse as sRed
  if (kh == 1) {
#pragma unroll
    for (int r = 0; r < 8; ++r) sRed[(wave - 4) * 256 + lane * 8 + r] = o[r];
  }
  __syncthreads();
  if (kh == 0) {
#pragma unroll
    for (int r = 0; r < 8; ++r) {
      const float tot = o[r] + sRed[wave * 256 + lane * 8 + r];
      const int j = jt + mt2 * 16 + hi8 + r;
      const size_t g = ((size_t)i * LDIM + j) * HCd + h * CHEAD + nt2 * 16 + n0;
      const float gate = (float)Gb[g];              // sigmoid already applied
      Ob[g] = (_Float16)(gate * tot);
    }
  }
}

// ---------------- kernel 5: dz = (G*out) @ Wout, masked ----------------
__global__ __launch_bounds__(256) void out_gemm_kernel(
    const _Float16* __restrict__ Ob, const _Float16* __restrict__ WoT,
    const float* __restrict__ pair_mask, float* __restrict__ dz) {
  extern __shared__ char smem[];
  _Float16* sA  = (_Float16*)smem;
  _Float16* sBT = sA + 128 * 128;
  const size_t m0 = (size_t)blockIdx.x * 128;
  const int tid = threadIdx.x;
  const _Float16* gA = Ob + m0 * HCd;
  for (int it = tid; it < 128 * 128 / 8; it += 256) {
    __builtin_prefetch(gA + 128 * 128 + it * 8, 0, 1);
    cp16(sA + it * 8, gA + it * 8);
    cp16(sBT + it * 8, WoT + it * 8);
  }
  wait_staged();
  const int wave = tid >> 5, lane = tid & 31;
  const int mrow = wave * 16;
  v8f acc[8];
#pragma unroll
  for (int nt = 0; nt < 8; ++nt) acc[nt] = zero8();
#pragma unroll
  for (int ks = 0; ks < 4; ++ks) {
    const v16h a = load_a_frag(sA + mrow * HCd + ks * 32, HCd, lane);
#pragma unroll
    for (int nt = 0; nt < 8; ++nt) {
      const v16h b = load_b_frag(sBT + (nt * 16) * HCd + ks * 32, HCd, lane);
      acc[nt] = wmma_f16(a, b, acc[nt]);
    }
  }
  const int hi8 = (lane & 16) ? 8 : 0;
  const int n0  = lane & 15;
#pragma unroll
  for (int nt = 0; nt < 8; ++nt) {
#pragma unroll
    for (int r = 0; r < 8; ++r) {
      const size_t row = m0 + mrow + hi8 + r;
      dz[row * CZd + nt * 16 + n0] = acc[nt][r] * pair_mask[row];
    }
  }
}

// ---------------- host launcher ----------------
extern "C" void kernel_launch(void* const* d_in, const int* in_sizes, int n_in,
                              void* d_out, int out_size, void* d_ws, size_t ws_size,
                              hipStream_t stream) {
  (void)in_sizes; (void)n_in; (void)out_size; (void)ws_size;
  const float* z         = (const float*)d_in[0];
  const float* pair_mask = (const float*)d_in[1];
  const unsigned char* res_mask = (const unsigned char*)d_in[2];
  const float* ln_gamma  = (const float*)d_in[3];
  const float* ln_beta   = (const float*)d_in[4];
  const float* Wq        = (const float*)d_in[5];
  const float* Wk        = (const float*)d_in[6];
  const float* Wv        = (const float*)d_in[7];
  const float* Wb        = (const float*)d_in[8];
  const float* Wg        = (const float*)d_in[9];
  const float* Wout      = (const float*)d_in[10];
  float* dz = (float*)d_out;

  char* ws = (char*)d_ws;
  size_t off = 0;
  auto take = [&](size_t n) { char* p = ws + off; off += (n + 255) & ~(size_t)255; return p; };
  const size_t szAct = (size_t)LL * HCd * sizeof(_Float16);   // 26.2 MB each
  _Float16* zln  = (_Float16*)take(szAct);
  _Float16* Qb   = (_Float16*)take(szAct);
  _Float16* Kb   = (_Float16*)take(szAct);
  _Float16* Vt   = (_Float16*)take(szAct);   // transposed V: [i][h][c][k]
  _Float16* Gb   = (_Float16*)take(szAct);
  _Float16* Obuf = (_Float16*)take(szAct);
  float*    Bb   = (float*)take((size_t)LL * NHEAD * sizeof(float));
  _Float16* WcT  = (_Float16*)take((size_t)5 * CZd * HCd * sizeof(_Float16));

  const int smemGemm = 2 * 128 * 128 * (int)sizeof(_Float16);                    // 65536
  const int smemAttn = (2 * LDIM * CHEAD + 32 * CHEAD + 32 * LDIM) * (int)sizeof(_Float16); // 63488

  wconv_kernel<<<dim3(64, 5, 1), 256, 0, stream>>>(Wq, Wk, Wv, Wg, Wout, WcT);
  layernorm_kernel<<<LL / 8, 256, 0, stream>>>(z, ln_gamma, ln_beta, zln);
  bias_kernel<<<LL, 128, 0, stream>>>(zln, Wb, Bb);
  proj_gemm_kernel<<<dim3(LL / 128, 4, 1), 256, smemGemm, stream>>>(zln, WcT, Qb, Kb, Vt, Gb);
  tri_attn_kernel<<<dim3(LDIM / 32, NHEAD, LDIM), 256, smemAttn, stream>>>(
      Qb, Kb, Vt, Gb, Bb, res_mask, Obuf);
  out_gemm_kernel<<<LL / 128, 256, smemGemm, stream>>>(Obuf, WcT + 4 * CZd * HCd, pair_mask, dz);
}
